// GaussianVoxelizer_82987358093357
// MI455X (gfx1250) — matrix operand order repaired and checked by reading, not compiled, not run on hardware
//
#include <hip/hip_runtime.h>

typedef __attribute__((ext_vector_type(16))) _Float16 v16h;
typedef __attribute__((ext_vector_type(8)))  float    v8f;
typedef __attribute__((ext_vector_type(2)))  float    v2f;

#define NGAUSS   1024
#define NCH      17
#define NVOX     640000
#define MAHA_T   7.5f
#define CHUNK    32            // gaussians per feature-WMMA K
#define NCHUNK   (NGAUSS / CHUNK)
#define NL2E     (-0.7213475204444817f)   // -0.5 * log2(e)

// LDS / staged workspace layout (bytes)
#define OFF_WA   0                         // 32c*6t*32lane*8B = 49152  (f32 maha A-operands, pre-swizzled)
#define OFF_B0   49152                     // 32c*32lane*16*2  = 32768  (f16 feature B, ch0..15)
#define OFF_B1   81920                     // 32c*32lane*16*2  = 32768  (f16 B: col0=ch16, col1=ones)
#define OFF_L2O  114688                    // 1024*4           = 4096   (log2(op))
#define TOTAL_BYTES 118784
#define OFF_GW   TOTAL_BYTES               // plain W[1024][12] f32 (ws only, not staged)

// ---------------- Kernel 1a: per-gaussian quadratic-form vector W ----------------
__global__ __launch_bounds__(256) void gv_prep(
    const float* __restrict__ means, const float* __restrict__ opac,
    const float* __restrict__ covs,
    float* __restrict__ gW,        // [G,12]
    float* __restrict__ gL2O)      // [G]
{
    int g = blockIdx.x * blockDim.x + threadIdx.x;
    if (g >= NGAUSS) return;

    const float* Cv = covs + g * 9;
    float c00 = Cv[0], c01 = Cv[1], c02 = Cv[2];
    float c11 = Cv[4], c12 = Cv[5], c22 = Cv[8];
    float a00 = c11 * c22 - c12 * c12;
    float a01 = c02 * c12 - c01 * c22;
    float a02 = c01 * c12 - c02 * c11;
    float a11 = c00 * c22 - c02 * c02;
    float a12 = c01 * c02 - c00 * c12;
    float a22 = c00 * c11 - c01 * c01;
    float det = c00 * a00 + c01 * a01 + c02 * a02;
    float id  = 1.0f / det;
    float i00 = a00 * id, i01 = a01 * id, i02 = a02 * id;
    float i11 = a11 * id, i12 = a12 * id, i22 = a22 * id;

    float mx = means[g * 3 + 0], my = means[g * 3 + 1], mz = means[g * 3 + 2];
    float b0 = -2.0f * (i00 * mx + i01 * my + i02 * mz);
    float b1 = -2.0f * (i01 * mx + i11 * my + i12 * mz);
    float b2 = -2.0f * (i02 * mx + i12 * my + i22 * mz);
    float cst = i00 * mx * mx + i11 * my * my + i22 * mz * mz
              + 2.0f * (i01 * mx * my + i02 * mx * mz + i12 * my * mz);

    float* w = gW + g * 12;
    w[0] = i00; w[1] = i11; w[2] = i22;
    w[3] = 2.0f * i01; w[4] = 2.0f * i02; w[5] = 2.0f * i12;
    w[6] = b0; w[7] = b1; w[8] = b2; w[9] = cst; w[10] = 0.f; w[11] = 0.f;

    gL2O[g] = log2f(opac[g]);
}

// ---------------- Kernel 1b: pre-swizzle f32 maha A-operands (16x4 f32 ISA layout) ----
__global__ __launch_bounds__(256) void gv_prep_swz(
    const float* __restrict__ gW,  // [G,12]
    float* __restrict__ gWA)       // [32c][6t][32lane][2]
{
    int t = blockIdx.x * blockDim.x + threadIdx.x;   // (chunk, lane)
    if (t >= NCHUNK * 32) return;
    int c = t >> 5, lane = t & 31;
    int h = (lane >> 4) & 1;
    #pragma unroll
    for (int rt = 0; rt < 2; ++rt) {
        int g = c * CHUNK + rt * 16 + (lane & 15);
        #pragma unroll
        for (int kt = 0; kt < 3; ++kt) {
            int idx = ((c * 6 + rt * 3 + kt) * 32 + lane) * 2;
            gWA[idx + 0] = gW[g * 12 + kt * 4 + 2 * h];
            gWA[idx + 1] = gW[g * 12 + kt * 4 + 1 + 2 * h];
        }
    }
}

// ---------------- Kernel 1c: pre-swizzle f16 feature B operands (32x16 f16 ISA layout) ----
// B1: column 0 = op*feat[16], column 1 = 1.0 (counts mask hits in the tensor pipe)
__global__ __launch_bounds__(256) void gv_prep_b(
    const float* __restrict__ opac, const float* __restrict__ feats,
    _Float16* __restrict__ gB0, _Float16* __restrict__ gB1)
{
    int t = blockIdx.x * blockDim.x + threadIdx.x;   // (chunk, lane)
    if (t >= NCHUNK * 32) return;
    int c = t >> 5, lane = t & 31;
    int ch = lane & 15, lgrp = lane & 16;
    #pragma unroll
    for (int e = 0; e < 16; ++e) {
        int g = c * CHUNK + (lgrp >> 1) + (e & 7) + ((e & 8) << 1);
        gB0[t * 16 + e] = (_Float16)(opac[g] * feats[g * NCH + ch]);
        _Float16 b1v = (_Float16)0.0f;
        if (ch == 0) b1v = (_Float16)(opac[g] * feats[g * NCH + 16]);
        if (ch == 1) b1v = (_Float16)1.0f;
        gB1[t * 16 + e] = b1v;
    }
}

// ---------------- Kernel 2: splat; maha via f32 WMMA, features+count via f16 WMMA ----------------
__global__ __launch_bounds__(256) void gv_splat(
    const float* __restrict__ coords,   // [N,3]
    const char*  __restrict__ gPack,    // TOTAL_BYTES of packed gaussian data
    float* __restrict__ outDens,        // [N]
    float* __restrict__ outFeat)        // [N,17]
{
    extern __shared__ char smem[];      // same layout as gPack, LDS offset 0
    const v2f*      sWA = (const v2f*)(smem + OFF_WA);
    const _Float16* sB0 = (const _Float16*)(smem + OFF_B0);
    const _Float16* sB1 = (const _Float16*)(smem + OFF_B1);
    const float*    sL  = (const float*)(smem + OFF_L2O);

    const int tid = threadIdx.x;

    // -------- async stage of all gaussian data into LDS --------
    {
        const int n16 = TOTAL_BYTES / 16;          // 7424 chunks of 16B
        for (int i = tid; i < n16; i += 256) {
            unsigned ldsoff = (unsigned)(i * 16);
            unsigned goff   = (unsigned)(i * 16);
            asm volatile(
                "global_load_async_to_lds_b128 %0, %1, %2"
                :: "v"(ldsoff), "v"(goff), "s"(gPack)
                : "memory");
        }
        asm volatile("s_wait_asynccnt 0" ::: "memory");
    }
    __syncthreads();

    const int lane  = tid & 31;
    const int wave  = tid >> 5;
    const int vbase = blockIdx.x * 128 + wave * 16;
    const int ch    = lane & 15;
    const int h     = (lane >> 4) & 1;

    // this lane's voxel (column N = lane&15 of the transposed maha product)
    const int v = vbase + ch;
    const float x = coords[v * 3 + 0];
    const float y = coords[v * 3 + 1];
    const float z = coords[v * 3 + 2];

    // U^T B-operands (4x16 f32 ISA layout), chunk-invariant:
    // U = [x2,y2,z2,xy,xz,yz,x,y,z,1,0,0]; VGPR0 = U[kt*4+2h], VGPR1 = U[kt*4+1+2h]
    v2f bu0, bu1, bu2;
    bu0.x = h ? (z * z) : (x * x);   bu0.y = h ? (x * y) : (y * y);
    bu1.x = h ? x       : (x * z);   bu1.y = h ? y       : (y * z);
    bu2.x = h ? 0.0f    : z;         bu2.y = h ? 0.0f    : 1.0f;

    v8f acc0 = {};   // feature channels 0..15
    v8f acc1 = {};   // col0 = channel 16, col1 = mask count
    v8f dva  = {};   // density partials, row-tile 0 (independent slots -> no serial chain)
    v8f dvb  = {};   // density partials, row-tile 1
    const v8f z8 = {};
    const _Float16 H1 = (_Float16)1.0f;
    const _Float16 H0 = (_Float16)0.0f;

    for (int c = 0; c < NCHUNK; ++c) {
        const int gbase = c * CHUNK;
        const int wab   = (c * 6) * 32 + lane;

        // ---- maha^T = W(16g x 12) x U^T(12 x 16v) via f32 WMMA, two gaussian row-tiles ----
        v8f m0 = __builtin_amdgcn_wmma_f32_16x16x4_f32(false, sWA[wab],        false, bu0, (short)0, z8, false, false);
        m0     = __builtin_amdgcn_wmma_f32_16x16x4_f32(false, sWA[wab + 32],   false, bu1, (short)0, m0, false, false);
        m0     = __builtin_amdgcn_wmma_f32_16x16x4_f32(false, sWA[wab + 64],   false, bu2, (short)0, m0, false, false);
        v8f m1 = __builtin_amdgcn_wmma_f32_16x16x4_f32(false, sWA[wab + 96],   false, bu0, (short)0, z8, false, false);
        m1     = __builtin_amdgcn_wmma_f32_16x16x4_f32(false, sWA[wab + 128],  false, bu1, (short)0, m1, false, false);
        m1     = __builtin_amdgcn_wmma_f32_16x16x4_f32(false, sWA[wab + 160],  false, bu2, (short)0, m1, false, false);

        // log2(op) for this lane's 16 gaussians (rows r+8h of each row-tile -> contiguous)
        v8f l2a = *(const v8f*)&sL[gbase + 8 * h];
        v8f l2b = *(const v8f*)&sL[gbase + 16 + 8 * h];

        // ---- per-pair: mask + weight; masks drop directly into f16 A-operand layout ----
        v16h amask;
        #pragma unroll
        for (int r = 0; r < 8; ++r) {
            float ma = m0[r];
            bool  sa = (ma <= MAHA_T);
            float wa = __builtin_amdgcn_exp2f(fmaf(ma, NL2E, l2a[r]));  // op*exp(-maha/2), raw v_exp_f32
            dva[r] += sa ? wa : 0.0f;
            amask[r] = sa ? H1 : H0;

            float mb = m1[r];
            bool  sb = (mb <= MAHA_T);
            float wb = __builtin_amdgcn_exp2f(fmaf(mb, NL2E, l2b[r]));
            dvb[r] += sb ? wb : 0.0f;
            amask[8 + r] = sb ? H1 : H0;
        }

        // ---- feature + count WMMAs: pre-swizzled B operands, single 32B vector loads ----
        v16h b0 = *(const v16h*)&sB0[(c * 32 + lane) * 16];
        v16h b1 = *(const v16h*)&sB1[(c * 32 + lane) * 16];
        acc0 = __builtin_amdgcn_wmma_f32_16x16x32_f16(
            false, amask, false, b0, (short)0, acc0, false, false);
        acc1 = __builtin_amdgcn_wmma_f32_16x16x32_f16(
            false, amask, false, b1, (short)0, acc1, false, false);
    }

    // ---- horizontal sum of density partials (tree), then fold lane halves ----
    float dacc;
    {
        v8f s = dva + dvb;
        float s0 = (s[0] + s[4]) + (s[1] + s[5]);
        float s1 = (s[2] + s[6]) + (s[3] + s[7]);
        dacc = s0 + s1;
    }
    float dFull = dacc + __shfl_xor(dacc, 16, 32);   // dsum for voxel row = lane&15
    const int hi     = h ? 8 : 0;
    const int cntSrc = h ? 17 : 1;                   // lane holding cnt for rows hi+0..7

    #pragma unroll
    for (int r = 0; r < 8; ++r) {
        float cnt_r = __shfl(acc1[r], cntSrc, 32);   // cnt[hi+r] from tensor-pipe count column
        float rdn   = __builtin_amdgcn_rcpf(fmaxf(cnt_r, 1.0f));
        int   vv    = vbase + hi + r;
        outFeat[vv * NCH + ch] = acc0[r] * rdn;
        if (ch == 0)
            outFeat[vv * NCH + 16] = acc1[r] * rdn;
        float dF_r = __shfl(dFull, hi + r, 32);      // dsum[hi+r]
        if (ch == 2)
            outDens[vv] = dF_r * rdn;
    }
}

extern "C" void kernel_launch(void* const* d_in, const int* in_sizes, int n_in,
                              void* d_out, int out_size, void* d_ws, size_t ws_size,
                              hipStream_t stream) {
    const float* coords = (const float*)d_in[0];  // [640000,3]
    const float* means  = (const float*)d_in[1];  // [1,1024,3]
    const float* opac   = (const float*)d_in[2];  // [1,1024,1]
    const float* feats  = (const float*)d_in[3];  // [1,1024,17]
    const float* covs   = (const float*)d_in[4];  // [1,1024,3,3]

    char*     base = (char*)d_ws;
    float*    gWA  = (float*)(base + OFF_WA);
    _Float16* gB0  = (_Float16*)(base + OFF_B0);
    _Float16* gB1  = (_Float16*)(base + OFF_B1);
    float*    gL2O = (float*)(base + OFF_L2O);
    float*    gW   = (float*)(base + OFF_GW);     // not staged to LDS

    float* outDens = (float*)d_out;          // [640000]
    float* outFeat = outDens + NVOX;         // [640000,17]

    gv_prep<<<(NGAUSS + 255) / 256, 256, 0, stream>>>(means, opac, covs, gW, gL2O);
    gv_prep_swz<<<(NCHUNK * 32 + 255) / 256, 256, 0, stream>>>(gW, gWA);
    gv_prep_b<<<(NCHUNK * 32 + 255) / 256, 256, 0, stream>>>(opac, feats, gB0, gB1);

    gv_splat<<<NVOX / 128, 256, (size_t)TOTAL_BYTES, stream>>>(
        coords, base, outDens, outFeat);
}
